// TransformerBlock_39127152066785
// MI455X (gfx1250) — compile-verified, matrix operand order checked by reading, not run on hardware
//
#include <hip/hip_runtime.h>
#include <stdint.h>
#include <stddef.h>

// ---------------- problem constants ----------------
#define S_LEN   2048
#define BATCH   2
#define D_MODEL 1024
#define NHEAD   16
#define HD      64
#define DFF_DIM 4096
#define NTOK    (S_LEN * BATCH)

typedef __attribute__((ext_vector_type(16))) __bf16 v16bf;
typedef __attribute__((ext_vector_type(8)))  float  v8f;

__device__ __forceinline__ uint16_t f2bf(float f) {
  uint32_t u = __float_as_uint(f);
  uint32_t r = u + 0x7FFFu + ((u >> 16) & 1u);   // round-to-nearest-even
  return (uint16_t)(r >> 16);
}

// ---- CDNA5 async global->LDS DMA path (ASYNCcnt), via inline asm ----
__device__ __forceinline__ void async_copy_b128(void* lds_dst, const void* src) {
  uint32_t lds_off = (uint32_t)(uintptr_t)lds_dst;   // low 32 bits of generic ptr = LDS offset
  asm volatile("global_load_async_to_lds_b128 %0, %1, off"
               :
               : "v"(lds_off), "v"(src)
               : "memory");
}
__device__ __forceinline__ void wait_async0() {
  asm volatile("s_wait_asynccnt 0" ::: "memory");
}

// ---------------- f32 -> bf16 convert ----------------
__global__ __launch_bounds__(256) void cvt_f32_bf16(const float* __restrict__ in,
                                                    uint16_t* __restrict__ out, int n) {
  int base = (blockIdx.x * 256 + threadIdx.x) * 4;
  if (base + 3 < n) {
    float4 v = *(const float4*)(in + base);
    out[base + 0] = f2bf(v.x);
    out[base + 1] = f2bf(v.y);
    out[base + 2] = f2bf(v.z);
    out[base + 3] = f2bf(v.w);
  }
}

// issue one 128x32 A tile + 64x32 B tile as async DMA into LDS
__device__ __forceinline__ void gemm_issue_tile(const uint16_t* A, const uint16_t* W,
                                                uint16_t* sAbuf, uint16_t* sBbuf,
                                                int tid, int rowBase, int colBase,
                                                int K, int k0) {
#pragma unroll
  for (int i = 0; i < 2; ++i) {                 // A tile: 512 x b128, 2 per thread
    int e = (tid + i * 256) * 8;
    int r = e >> 5, cc = e & 31;
    async_copy_b128(&sAbuf[r * 32 + cc], &A[(size_t)(rowBase + r) * K + k0 + cc]);
  }
  {                                             // B tile: 256 x b128, 1 per thread
    int e = tid * 8;
    int r = e >> 5, cc = e & 31;
    async_copy_b128(&sBbuf[r * 32 + cc], &W[(size_t)(colBase + r) * K + k0 + cc]);
  }
}

// ---------------- WMMA GEMM:  C[M,N] = A[M,K](bf16,row) * W[N,K](bf16,row)^T + bias ---------
// MODE 0: out bf16 to [B,H,S,HD] layout, scaled (QKV projections)
// MODE 2: outF f32 = C + resid                         (O-proj / FFN2 residual)
// MODE 3: outB bf16 = gelu_tanh(C)  row-major [M,N]    (FFN1)
template <int MODE>
__global__ __launch_bounds__(256) void gemm_bf16_kernel(
    const uint16_t* __restrict__ A, const uint16_t* __restrict__ W,
    const float* __restrict__ bias, const float* __restrict__ resid,
    float* __restrict__ outF, uint16_t* __restrict__ outB,
    int M, int N, int K, float scale) {
  __shared__ __attribute__((aligned(32))) uint16_t sA[2][128 * 32];  // 2 x 8 KB
  __shared__ __attribute__((aligned(32))) uint16_t sB[2][64 * 32];   // 2 x 4 KB

  const int tid  = threadIdx.x;
  const int lane = tid & 31;
  const int wave = tid >> 5;
  const int wm = wave & 3;        // 4 waves along M
  const int wn = wave >> 2;       // 2 waves along N
  const int rowBase = blockIdx.y * 128;
  const int colBase = blockIdx.x * 64;
  const int m0   = lane & 15;
  const int koff = (lane >> 4) * 16;

  v8f acc[2][2];
#pragma unroll
  for (int mi = 0; mi < 2; ++mi)
#pragma unroll
    for (int ni = 0; ni < 2; ++ni)
#pragma unroll
      for (int j = 0; j < 8; ++j) acc[mi][ni][j] = 0.0f;

  const int ksteps = K >> 5;
  gemm_issue_tile(A, W, sA[0], sB[0], tid, rowBase, colBase, K, 0);

  for (int kt = 0; kt < ksteps; ++kt) {
    wait_async0();        // my DMA done ...
    __syncthreads();      // ... everyone's DMA done -> tile visible
    if (kt + 1 < ksteps)  // overlap next tile's DMA with this tile's WMMA
      gemm_issue_tile(A, W, sA[(kt + 1) & 1], sB[(kt + 1) & 1],
                      tid, rowBase, colBase, K, (kt + 1) << 5);

    const uint16_t* cA = sA[kt & 1];
    const uint16_t* cB = sB[kt & 1];
    v16bf afr[2], bfr[2];
#pragma unroll
    for (int mi = 0; mi < 2; ++mi)
      afr[mi] = *(const v16bf*)(&cA[(wm * 32 + mi * 16 + m0) * 32 + koff]);
#pragma unroll
    for (int ni = 0; ni < 2; ++ni)
      bfr[ni] = *(const v16bf*)(&cB[(wn * 32 + ni * 16 + m0) * 32 + koff]);
#pragma unroll
    for (int mi = 0; mi < 2; ++mi)
#pragma unroll
      for (int ni = 0; ni < 2; ++ni)
        acc[mi][ni] = __builtin_amdgcn_wmma_f32_16x16x32_bf16(
            false, afr[mi], false, bfr[ni], (short)0, acc[mi][ni], false, false);
  }

#pragma unroll
  for (int mi = 0; mi < 2; ++mi)
#pragma unroll
    for (int ni = 0; ni < 2; ++ni)
#pragma unroll
      for (int j = 0; j < 8; ++j) {
        int rl = wm * 32 + mi * 16 + j + ((lane >> 4) << 3);
        int cl = wn * 32 + ni * 16 + (lane & 15);
        int gm = rowBase + rl;
        int gn = colBase + cl;
        float v = acc[mi][ni][j] + bias[gn];
        if (MODE == 0) {
          v *= scale;
          int s = gm / BATCH, b = gm % BATCH;
          int h = gn >> 6, dd = gn & 63;
          outB[(((size_t)(b * NHEAD + h)) * S_LEN + s) * HD + dd] = f2bf(v);
        } else if (MODE == 2) {
          outF[(size_t)gm * N + gn] = v + resid[(size_t)gm * N + gn];
        } else if (MODE == 3) {
          float t  = 0.7978845608028654f * (v + 0.044715f * v * v * v);
          float ge = 0.5f * v * (1.0f + tanhf(t));
          outB[(size_t)gm * N + gn] = f2bf(ge);
        }
      }
}

// ---------------- flash attention (online softmax, WMMA for QK^T and PV) --------------
// grid: (S/64, B*H), block 128 (4 waves); each wave owns 16 query rows.
__global__ __launch_bounds__(128) void flash_attn_kernel(
    const uint16_t* __restrict__ Q, const uint16_t* __restrict__ Kg,
    const uint16_t* __restrict__ Vg, uint16_t* __restrict__ O) {
  __shared__ __attribute__((aligned(32))) uint16_t sK[64 * 64];    // 8 KB, [key][d]
  __shared__ __attribute__((aligned(32))) uint16_t sVT[64 * 64];   // 8 KB, [d][key]
  __shared__ __attribute__((aligned(32))) uint16_t sP[4 * 16 * 64];// 8 KB, per-wave P

  const int tid  = threadIdx.x;
  const int lane = tid & 31;
  const int wave = tid >> 5;
  const int bh   = blockIdx.y;
  const int b    = bh >> 4;
  const int h    = bh & 15;
  const int qt   = blockIdx.x;
  const size_t headBase = (size_t)bh * S_LEN * HD;
  const int m0   = lane & 15;
  const int koff = (lane >> 4) * 16;

  // Q fragments for this wave's 16 rows (scale already folded into Q)
  v16bf aq[2];
#pragma unroll
  for (int kk = 0; kk < 2; ++kk)
    aq[kk] = *(const v16bf*)(
        &Q[headBase + (size_t)(qt * 64 + wave * 16 + m0) * HD + kk * 32 + koff]);

  v8f sc[4], ov[4];
  float rmax[8], rl[8];
#pragma unroll
  for (int nt = 0; nt < 4; ++nt)
#pragma unroll
    for (int j = 0; j < 8; ++j) ov[nt][j] = 0.0f;
#pragma unroll
  for (int j = 0; j < 8; ++j) { rmax[j] = -1e30f; rl[j] = 0.0f; }

  for (int kt = 0; kt < S_LEN / 64; ++kt) {
    __syncthreads();  // previous-iteration readers done
    // K tile row-major via async DMA to LDS
#pragma unroll
    for (int i = 0; i < 4; ++i) {
      int e = (tid + i * 128) * 8;
      int r = e >> 6, cc = e & 63;
      async_copy_b128(&sK[r * 64 + cc],
                      &Kg[headBase + (size_t)(kt * 64 + r) * HD + cc]);
    }
    // V tile transposed into LDS ([d][key]) so PV B-fragments are contiguous
#pragma unroll
    for (int i = 0; i < 4; ++i) {
      int e = tid * 32 + i * 8;
      uint4 v = *(const uint4*)(&Vg[headBase + (size_t)kt * 64 * HD + e]);
      const uint16_t* pv = (const uint16_t*)&v;
#pragma unroll
      for (int j = 0; j < 8; ++j) {
        int idx = e + j;
        sVT[(idx & 63) * 64 + (idx >> 6)] = pv[j];
      }
    }
    wait_async0();
    __syncthreads();

    // scores: 16x64 per wave = 4 n-tiles x 2 k-steps of WMMA
#pragma unroll
    for (int nt = 0; nt < 4; ++nt) {
#pragma unroll
      for (int j = 0; j < 8; ++j) sc[nt][j] = 0.0f;
#pragma unroll
      for (int kk = 0; kk < 2; ++kk) {
        v16bf bk = *(const v16bf*)(&sK[(nt * 16 + m0) * 64 + kk * 32 + koff]);
        sc[nt] = __builtin_amdgcn_wmma_f32_16x16x32_bf16(
            false, aq[kk], false, bk, (short)0, sc[nt], false, false);
      }
    }

    // online softmax (rows live across 16-lane groups in the C layout)
#pragma unroll
    for (int j = 0; j < 8; ++j) {
      float mj = fmaxf(fmaxf(sc[0][j], sc[1][j]), fmaxf(sc[2][j], sc[3][j]));
#pragma unroll
      for (int off = 1; off < 16; off <<= 1) mj = fmaxf(mj, __shfl_xor(mj, off, 32));
      float mnew = fmaxf(rmax[j], mj);
      float corr = __expf(rmax[j] - mnew);
      rmax[j] = mnew;
      float ps = 0.0f;
#pragma unroll
      for (int nt = 0; nt < 4; ++nt) {
        float p = __expf(sc[nt][j] - mnew);
        sc[nt][j] = p;
        ps += p;
      }
#pragma unroll
      for (int off = 1; off < 16; off <<= 1) ps += __shfl_xor(ps, off, 32);
      rl[j] = rl[j] * corr + ps;
#pragma unroll
      for (int nt = 0; nt < 4; ++nt) ov[nt][j] *= corr;
    }

    // stage P (C-layout -> A-layout) through per-wave LDS region
#pragma unroll
    for (int nt = 0; nt < 4; ++nt)
#pragma unroll
      for (int j = 0; j < 8; ++j) {
        int m = j + ((lane >> 4) << 3);
        sP[wave * 1024 + m * 64 + nt * 16 + m0] = f2bf(sc[nt][j]);
      }
    __syncthreads();

    // PV: accumulate 16x64 output
#pragma unroll
    for (int nt = 0; nt < 4; ++nt)
#pragma unroll
      for (int kk = 0; kk < 2; ++kk) {
        v16bf ap = *(const v16bf*)(&sP[wave * 1024 + m0 * 64 + kk * 32 + koff]);
        v16bf bv = *(const v16bf*)(&sVT[(nt * 16 + m0) * 64 + kk * 32 + koff]);
        ov[nt] = __builtin_amdgcn_wmma_f32_16x16x32_bf16(
            false, ap, false, bv, (short)0, ov[nt], false, false);
      }
  }

  // normalize and scatter to [token, D] bf16 (for O-projection A operand)
#pragma unroll
  for (int j = 0; j < 8; ++j) {
    float inv = 1.0f / rl[j];
    int sg = qt * 64 + wave * 16 + j + ((lane >> 4) << 3);
    size_t tok = (size_t)sg * BATCH + b;
#pragma unroll
    for (int nt = 0; nt < 4; ++nt)
      O[tok * D_MODEL + h * 64 + nt * 16 + m0] = f2bf(ov[nt][j] * inv);
  }
}

// ---------------- LayerNorm (one row per block), dual f32/bf16 output --------------
__global__ __launch_bounds__(256) void ln_kernel(const float* __restrict__ in,
                                                 const float* __restrict__ g,
                                                 const float* __restrict__ be,
                                                 float* __restrict__ outF,
                                                 uint16_t* __restrict__ outB) {
  __shared__ float rs[256], rq[256];
  const int tid = threadIdx.x;
  const size_t row = blockIdx.x;
  const float* p = in + row * D_MODEL;
  float s = 0.0f, q = 0.0f;
#pragma unroll
  for (int i = 0; i < 4; ++i) {
    float v = p[tid + i * 256];
    s += v;
    q += v * v;
  }
  rs[tid] = s;
  rq[tid] = q;
  __syncthreads();
  for (int st = 128; st > 0; st >>= 1) {
    if (tid < st) { rs[tid] += rs[tid + st]; rq[tid] += rq[tid + st]; }
    __syncthreads();
  }
  float mu  = rs[0] * (1.0f / D_MODEL);
  float var = rq[0] * (1.0f / D_MODEL) - mu * mu;
  float inv = rsqrtf(var + 1e-5f);
#pragma unroll
  for (int i = 0; i < 4; ++i) {
    int idx = tid + i * 256;
    float v = (p[idx] - mu) * inv * g[idx] + be[idx];
    outF[row * D_MODEL + idx] = v;
    if (outB) outB[row * D_MODEL + idx] = f2bf(v);
  }
}

// ---------------- host orchestration ----------------
extern "C" void kernel_launch(void* const* d_in, const int* in_sizes, int n_in,
                              void* d_out, int out_size, void* d_ws, size_t ws_size,
                              hipStream_t stream) {
  (void)in_sizes; (void)n_in; (void)out_size; (void)ws_size;
  const float* x   = (const float*)d_in[0];
  const float* Wq  = (const float*)d_in[1];
  const float* bq  = (const float*)d_in[2];
  const float* Wk  = (const float*)d_in[3];
  const float* bk  = (const float*)d_in[4];
  const float* Wv  = (const float*)d_in[5];
  const float* bv  = (const float*)d_in[6];
  const float* Wo  = (const float*)d_in[7];
  const float* bo  = (const float*)d_in[8];
  const float* W1  = (const float*)d_in[9];
  const float* b1  = (const float*)d_in[10];
  const float* W2  = (const float*)d_in[11];
  const float* b2  = (const float*)d_in[12];
  const float* g1  = (const float*)d_in[13];
  const float* be1 = (const float*)d_in[14];
  const float* g2  = (const float*)d_in[15];
  const float* be2 = (const float*)d_in[16];
  float* out = (float*)d_out;

  char* ws = (char*)d_ws;
  const size_t MB = 1ull << 20;
  uint16_t* xb    = (uint16_t*)(ws + 0 * MB);     // 8 MB  [NTOK,D] bf16
  uint16_t* qb    = (uint16_t*)(ws + 8 * MB);     // 8 MB  [B,H,S,HD]
  uint16_t* kb    = (uint16_t*)(ws + 16 * MB);    // 8 MB
  uint16_t* vb    = (uint16_t*)(ws + 24 * MB);    // 8 MB
  uint16_t* attnb = (uint16_t*)(ws + 32 * MB);    // 8 MB  [NTOK,D]
  float*    res1  = (float*)   (ws + 40 * MB);    // 16 MB [NTOK,D] f32 (reused for res2)
  float*    xn1f  = (float*)   (ws + 56 * MB);    // 16 MB
  uint16_t* xn1b  = (uint16_t*)(ws + 72 * MB);    // 8 MB
  uint16_t* yb    = (uint16_t*)(ws + 80 * MB);    // 32 MB [NTOK,DFF]
  uint16_t* Wqb   = (uint16_t*)(ws + 112 * MB);   // 2 MB each
  uint16_t* Wkb   = (uint16_t*)(ws + 114 * MB);
  uint16_t* Wvb   = (uint16_t*)(ws + 116 * MB);
  uint16_t* Wob   = (uint16_t*)(ws + 118 * MB);
  uint16_t* W1b   = (uint16_t*)(ws + 120 * MB);   // 8 MB
  uint16_t* W2b   = (uint16_t*)(ws + 128 * MB);   // 8 MB  (total 136 MB)

  const int nXD = NTOK * D_MODEL;                 // 4M
  const int nDD = D_MODEL * D_MODEL;              // 1M
  const int nDF = D_MODEL * DFF_DIM;              // 4M
  cvt_f32_bf16<<<nXD / 1024, 256, 0, stream>>>(x,  xb,  nXD);
  cvt_f32_bf16<<<nDD / 1024, 256, 0, stream>>>(Wq, Wqb, nDD);
  cvt_f32_bf16<<<nDD / 1024, 256, 0, stream>>>(Wk, Wkb, nDD);
  cvt_f32_bf16<<<nDD / 1024, 256, 0, stream>>>(Wv, Wvb, nDD);
  cvt_f32_bf16<<<nDD / 1024, 256, 0, stream>>>(Wo, Wob, nDD);
  cvt_f32_bf16<<<nDF / 1024, 256, 0, stream>>>(W1, W1b, nDF);
  cvt_f32_bf16<<<nDF / 1024, 256, 0, stream>>>(W2, W2b, nDF);

  // QKV projections (attention scale 1/sqrt(64) folded into Q)
  dim3 gP(D_MODEL / 64, NTOK / 128);
  gemm_bf16_kernel<0><<<gP, 256, 0, stream>>>(xb, Wqb, bq, nullptr, nullptr, qb,
                                              NTOK, D_MODEL, D_MODEL, 0.125f);
  gemm_bf16_kernel<0><<<gP, 256, 0, stream>>>(xb, Wkb, bk, nullptr, nullptr, kb,
                                              NTOK, D_MODEL, D_MODEL, 1.0f);
  gemm_bf16_kernel<0><<<gP, 256, 0, stream>>>(xb, Wvb, bv, nullptr, nullptr, vb,
                                              NTOK, D_MODEL, D_MODEL, 1.0f);

  // flash attention
  dim3 gA(S_LEN / 64, BATCH * NHEAD);
  flash_attn_kernel<<<gA, 128, 0, stream>>>(qb, kb, vb, attnb);

  // output projection + residual (f32)
  gemm_bf16_kernel<2><<<gP, 256, 0, stream>>>(attnb, Wob, bo, x, res1, nullptr,
                                              NTOK, D_MODEL, D_MODEL, 1.0f);

  // layernorm 1 -> f32 (for residual) and bf16 (for FFN A operand)
  ln_kernel<<<NTOK, 256, 0, stream>>>(res1, g1, be1, xn1f, xn1b);

  // FFN up + GELU -> bf16
  dim3 gF1(DFF_DIM / 64, NTOK / 128);
  gemm_bf16_kernel<3><<<gF1, 256, 0, stream>>>(xn1b, W1b, b1, nullptr, nullptr, yb,
                                               NTOK, DFF_DIM, D_MODEL, 1.0f);

  // FFN down + residual (reuse res1 buffer)
  gemm_bf16_kernel<2><<<gP, 256, 0, stream>>>(yb, W2b, b2, xn1f, res1, nullptr,
                                              NTOK, D_MODEL, DFF_DIM, 1.0f);

  // layernorm 2 -> final f32 output
  ln_kernel<<<NTOK, 256, 0, stream>>>(res1, g2, be2, out, nullptr);
}
